// DynamicMultiRNN_52553219834056
// MI455X (gfx1250) — matrix operand order repaired and checked by reading, not compiled, not run on hardware
//
#include <hip/hip_runtime.h>
#include <hip/hip_bf16.h>

typedef __attribute__((ext_vector_type(16))) __bf16 v16bf;
typedef __attribute__((ext_vector_type(8)))  __bf16 v8bf;
typedef __attribute__((ext_vector_type(8)))  float  v8f;

static constexpr int BATCH  = 1024;
static constexpr int TSTEPS = 200;
static constexpr int DDIM   = 128;   // D
static constexpr int ND4    = 512;   // 4*D
static constexpr int KCAT   = 256;   // K = D (input) + D (hidden)
static constexpr int KP     = 264;   // padded K pitch: 264 == 8 (mod 128) -> conflict-free b128 frags
static constexpr int BM     = 16;    // batch rows per workgroup
static constexpr int OSP    = 132;   // out-staging row pitch (floats), pads banks

__device__ __forceinline__ float sigmoidf_(float x) {
    return 1.0f / (1.0f + __expf(-x));
}

// A(16xKCAT bf16) x WT(KCATx512 bf16, N-major): wave `wave` computes the four
// gate tiles {g*8 + wave | g=0..3} (N base g*128 + wave*16), K=256 -> 8 WMMAs
// each; accumulators stay in registers (acc[g]).
__device__ __forceinline__ void gemm_gates(const __bf16* __restrict__ Ab,
                                           const __bf16* __restrict__ WT,
                                           v8f acc[4], int wave, int lane)
{
    const int m  = lane & 15;
    const int hi = lane >> 4;
    // A frags: lanes0-15 K {0..7,16..23}, lanes16-31 K {8..15,24..31} per 16x32 tile
    v16bf afrag[8];
    const __bf16* Arow = Ab + m * KP + (hi ? 8 : 0);
    #pragma unroll
    for (int kb = 0; kb < 8; ++kb) {
        v8bf lo = *(const v8bf*)(Arow + kb * 32);
        v8bf h2 = *(const v8bf*)(Arow + kb * 32 + 16);
        afrag[kb] = __builtin_shufflevector(lo, h2,
            0,1,2,3,4,5,6,7,8,9,10,11,12,13,14,15);
    }
    #pragma unroll
    for (int g = 0; g < 4; ++g) {
        const __bf16* Brow = WT + (g * DDIM + wave * 16 + m) * KP + (hi ? 8 : 0);
        v8f a = {0.f, 0.f, 0.f, 0.f, 0.f, 0.f, 0.f, 0.f};
        #pragma unroll
        for (int kb = 0; kb < 8; ++kb) {
            v8bf blo = *(const v8bf*)(Brow + kb * 32);
            v8bf bhi = *(const v8bf*)(Brow + kb * 32 + 16);
            v16bf bfrag = __builtin_shufflevector(blo, bhi,
                0,1,2,3,4,5,6,7,8,9,10,11,12,13,14,15);
            a = __builtin_amdgcn_wmma_f32_16x16x32_bf16(
                false, afrag[kb], false, bfrag, (short)0, a, false, false);
        }
        acc[g] = a;
    }
}

__global__ __launch_bounds__(256)
void lstm2_residual_wmma_kernel(const float* __restrict__ x,
                                const float* __restrict__ W,
                                const float* __restrict__ U,
                                const float* __restrict__ b,
                                float* __restrict__ out)
{
    extern __shared__ __align__(16) char smem[];
    __bf16* WT   = (__bf16*)smem;              // [512][KP]  [W;U]^T bf16
    __bf16* A1   = WT + ND4 * KP;              // [16][KP]   layer1 A = [x_t | h1_prev]
    __bf16* A2   = A1 + BM * KP;               // [16][KP]   layer2 A = [h1 | h2_prev]
    float*  osta = (float*)(A2 + BM * KP);     // [16][OSP]  out staging (coalesce)

    const int tid  = threadIdx.x;
    const int lane = tid & 31;
    const int wave = tid >> 5;
    const int b0   = blockIdx.x * BM;

    // ---- one-time: weights transposed to bf16, N-major with padded K pitch
    for (int idx = tid; idx < ND4 * KCAT; idx += 256) {
        const int n = idx & (ND4 - 1);     // coalesced global reads over n
        const int k = idx >> 9;
        const float v = (k < DDIM) ? W[k * ND4 + n] : U[(k - DDIM) * ND4 + n];
        WT[n * KP + k] = (__bf16)v;
    }
    for (int i = tid; i < BM * KP; i += 256) {   // h(-1) = 0
        A1[i] = (__bf16)0.0f;
        A2[i] = (__bf16)0.0f;
    }

    // ---- per-lane constants in C-tile layout: this lane owns column
    //      colg of every gate, rows r+8*hi (r=0..7)
    const int m    = lane & 15;
    const int hi   = lane >> 4;
    const int colg = wave * 16 + m;              // 0..127
    const float bi_ = b[colg];
    const float bf_ = b[DDIM + colg];
    const float bg_ = b[2 * DDIM + colg];
    const float bo_ = b[3 * DDIM + colg];

    float c1r[8], c2r[8], h1r[8];
    #pragma unroll
    for (int r = 0; r < 8; ++r) { c1r[r] = 0.f; c2r[r] = 0.f; h1r[r] = 0.f; }

    // thread-linear mapping for x staging / output copy
    const int rowL  = tid >> 4;          // 0..15
    const int colbL = (tid & 15) * 8;    // 0..120

    __syncthreads();

    for (int t = 0; t < TSTEPS; ++t) {
        // ---- stage x_t (bf16) into A1[:, 0:128]
        {
            const float* xp = x + ((size_t)(b0 + rowL) * TSTEPS + t) * DDIM + colbL;
            float4 lo = *(const float4*)xp;
            float4 hv = *(const float4*)(xp + 4);
            __bf16* dst = A1 + rowL * KP + colbL;
            dst[0] = (__bf16)lo.x; dst[1] = (__bf16)lo.y;
            dst[2] = (__bf16)lo.z; dst[3] = (__bf16)lo.w;
            dst[4] = (__bf16)hv.x; dst[5] = (__bf16)hv.y;
            dst[6] = (__bf16)hv.z; dst[7] = (__bf16)hv.w;
            if (t + 1 < TSTEPS) __builtin_prefetch(xp + DDIM, 0, 0);
        }
        __syncthreads();

        // ===== layer 1: z1 = [x_t | h1] @ [W;U]^T  (accumulators in registers)
        v8f acc[4];
        gemm_gates(A1, WT, acc, wave, lane);
        __syncthreads();   // all waves done reading A1/A2 h-halves before overwrite

        // ---- layer-1 gates in registers; h1 -> A1 hi half & A2 lo half
        #pragma unroll
        for (int r = 0; r < 8; ++r) {
            const float zi = acc[0][r] + bi_;
            const float zf = acc[1][r] + bf_;
            const float zg = acc[2][r] + bg_;
            const float zo = acc[3][r] + bo_;
            const float cn = sigmoidf_(zf) * c1r[r] + sigmoidf_(zi) * tanhf(zg);
            const float hn = sigmoidf_(zo) * tanhf(cn);
            c1r[r] = cn;
            h1r[r] = hn;
            const int row = r + 8 * hi;
            const __bf16 hb = (__bf16)hn;
            A1[row * KP + DDIM + colg] = hb;   // h1_prev for step t+1
            A2[row * KP + colg]        = hb;   // layer-2 input
        }
        __syncthreads();

        // ===== layer 2 (shared weights): z2 = [h1 | h2] @ [W;U]^T
        gemm_gates(A2, WT, acc, wave, lane);
        __syncthreads();   // all waves done reading A2 hi half before overwrite

        // ---- layer-2 gates + residual; h2 -> A2 hi half & out staging
        #pragma unroll
        for (int r = 0; r < 8; ++r) {
            const float zi = acc[0][r] + bi_;
            const float zf = acc[1][r] + bf_;
            const float zg = acc[2][r] + bg_;
            const float zo = acc[3][r] + bo_;
            const float cn = sigmoidf_(zf) * c2r[r] + sigmoidf_(zi) * tanhf(zg);
            const float hn = sigmoidf_(zo) * tanhf(cn) + h1r[r];   // residual
            c2r[r] = cn;
            const int row = r + 8 * hi;
            A2[row * KP + DDIM + colg] = (__bf16)hn;  // h2_prev for step t+1
            osta[row * OSP + colg]     = hn;
        }
        __syncthreads();

        // ---- coalesced output: staging -> out[b, t, :]
        {
            const float* src = osta + rowL * OSP + colbL;
            float4 lo = *(const float4*)src;
            float4 hv = *(const float4*)(src + 4);
            float* orow = out + ((size_t)(b0 + rowL) * TSTEPS + t) * DDIM + colbL;
            *(float4*)orow       = lo;
            *(float4*)(orow + 4) = hv;
        }
        // next iteration's x-stage touches only A1 lo; out-stage rewrite is
        // separated by >=2 barriers -> no barrier needed here
    }
}

extern "C" void kernel_launch(void* const* d_in, const int* in_sizes, int n_in,
                              void* d_out, int out_size, void* d_ws, size_t ws_size,
                              hipStream_t stream) {
    (void)in_sizes; (void)n_in; (void)out_size; (void)d_ws; (void)ws_size;
    const float* x = (const float*)d_in[0];
    const float* W = (const float*)d_in[1];
    const float* U = (const float*)d_in[2];
    const float* b = (const float*)d_in[3];
    // d_in[4] = seq_len: ignored (reference loop runs full T)
    float* out = (float*)d_out;

    constexpr size_t SMEM =
        (size_t)(ND4 * KP + 2 * BM * KP) * sizeof(__bf16) +
        (size_t)(BM * OSP) * sizeof(float);     // 295,680 B < 320 KB WGP LDS

    hipFuncSetAttribute((const void*)lstm2_residual_wmma_kernel,
                        hipFuncAttributeMaxDynamicSharedMemorySize, (int)SMEM);

    lstm2_residual_wmma_kernel<<<BATCH / BM, 256, SMEM, stream>>>(x, W, U, b, out);
}
